// RACE_42631845380938
// MI455X (gfx1250) — matrix operand order repaired and checked by reading, not compiled, not run on hardware
//
#include <hip/hip_runtime.h>
#include <hip/hip_bf16.h>
#include <math.h>

// ---------------- problem constants (match reference) ----------------
#define N_NODES   20000
#define G_GRAPHS  8
#define NP_NODES  2500
#define E_EDGES   640000
#define C_CH      64
#define S_SP      8
#define NB_B      8
#define L_LAYERS  3
#define CUTF      6.0f
#define AVGF      32.0f

typedef __attribute__((ext_vector_type(16))) _Float16 v16h;
typedef __attribute__((ext_vector_type(8)))  _Float16 v8h;
typedef __attribute__((ext_vector_type(8)))  float    v8f;

// ---------------- WMMA helpers (CDNA5 gfx1250, wave32) ----------------
__device__ __forceinline__ v8f wmma16(v16h a, v16h b, v8f c) {
  // D = A(16x32 f16) * B(32x16 f16) + C(16x16 f32)
  return __builtin_amdgcn_wmma_f32_16x16x32_f16(false, a, false, b, (short)0, c, false, false);
}

__device__ __forceinline__ float silu_f(float x) { return x / (1.0f + __expf(-x)); }

// B fragment from ROW-major B[k][n]: lane holds col n = n0+lane%16, rows k = 16*(lane/16)+e.
// (strided; used only for once-per-kernel setup loads)
__device__ __forceinline__ v16h load_bfrag(const _Float16* B, int ld, int n0, int lane) {
  const int half = lane >> 4;
  const int n = n0 + (lane & 15);
  v16h f;
#pragma unroll
  for (int e = 0; e < 16; ++e) f[e] = B[(16 * half + e) * ld + n];
  return f;
}

// B fragment from TRANSPOSED storage Bt[n][k] (ldk = k-stride): 16 contiguous halves
// per lane -> compiler emits 2x ds_load_b128.
__device__ __forceinline__ v16h load_bfrag_t(const _Float16* Bt, int ldk, int n0, int k0,
                                             int lane) {
  const int base = (n0 + (lane & 15)) * ldk + k0 + 16 * (lane >> 4);
  v16h f;
#pragma unroll
  for (int e = 0; e < 16; ++e) f[e] = Bt[base + e];
  return f;
}

// A fragment from row-major A[m][k]: lane holds row m = lane%16,
// cols k = k0 + 16*(lane/16) + e  (32B-contiguous run -> 2x ds_load_b128)
__device__ __forceinline__ v16h load_afrag(const _Float16* A, int ld, int k0, int lane) {
  const int m = lane & 15, half = lane >> 4;
  v16h f;
#pragma unroll
  for (int e = 0; e < 16; ++e) {
    int k = k0 + 16 * half + e;
    f[e] = A[m * ld + k];
  }
  return f;
}

// ---------------- small utility kernels ----------------
__global__ void k_zero(float* __restrict__ p, int count) {
  for (int i = blockIdx.x * blockDim.x + threadIdx.x; i < count; i += gridDim.x * blockDim.x)
    p[i] = 0.0f;
}

// M2[sp,o] = sum_c W_embed[sp,c] * W_x[c,o]   (one-hot @ W_embed @ W_x collapsed)
__global__ void k_m2(const float* __restrict__ We, const float* __restrict__ Wx,
                     float* __restrict__ M2) {
  int t = threadIdx.x;
  if (t >= S_SP * C_CH) return;
  int sp = t >> 6, o = t & 63;
  float acc = 0.0f;
  for (int c = 0; c < 64; ++c) acc += We[sp * 64 + c] * Wx[c * 64 + o];
  M2[t] = acc;
}

// s = W_embed[species], xf = M2[species], v = 0
__global__ void k_embed(const int* __restrict__ species, const float* __restrict__ We,
                        const float* __restrict__ M2, float* __restrict__ s,
                        float* __restrict__ xf, float* __restrict__ v) {
  int id = blockIdx.x * blockDim.x + threadIdx.x;
  if (id >= N_NODES * 64) return;
  int n = id >> 6, c = id & 63;
  int sp = species[n];
  s[id] = We[sp * 64 + c];
  xf[id] = M2[sp * 64 + c];
  v[(size_t)id * 3 + 0] = 0.0f;
  v[(size_t)id * 3 + 1] = 0.0f;
  v[(size_t)id * 3 + 2] = 0.0f;
}

// ---------------- edge geometry: y1, mask, bessel*envelope (f16) ----------------
__global__ void k_edge_geom(const float* __restrict__ pos, const float* __restrict__ cell,
                            const float* __restrict__ Sij, const int* __restrict__ snd,
                            const int* __restrict__ rcv, const int* __restrict__ batch,
                            _Float16* __restrict__ ef16, float* __restrict__ y1,
                            float* __restrict__ maskA) {
  int e = blockIdx.x * blockDim.x + threadIdx.x;
  if (e >= E_EDGES) return;
  int sI = snd[e], rI = rcv[e];
  int g = batch[sI];
  const float* Cg = cell + (size_t)g * 9;
  float S0 = Sij[(size_t)e * 3 + 0], S1 = Sij[(size_t)e * 3 + 1], S2 = Sij[(size_t)e * 3 + 2];
  float R[3];
#pragma unroll
  for (int j = 0; j < 3; ++j) {
    float sh = S0 * Cg[0 * 3 + j] + S1 * Cg[1 * 3 + j] + S2 * Cg[2 * 3 + j];
    R[j] = (pos[(size_t)rI * 3 + j] - pos[(size_t)sI * 3 + j] + sh) * (1.0f / CUTF);
  }
  float len = sqrtf(R[0] * R[0] + R[1] * R[1] + R[2] * R[2]);
  float mk = (len > 0.0f) ? 1.0f : 0.0f;
  float safe = (len > 0.0f) ? len : 1.0f;
  float inv = 1.0f / safe;
  const float SQ3 = 1.7320508075688772f;
#pragma unroll
  for (int j = 0; j < 3; ++j) y1[(size_t)e * 3 + j] = SQ3 * R[j] * inv;
  float x = safe;
  float env = (x < 1.0f) ? (1.0f - 6.0f * x * x + 8.0f * x * x * x - 3.0f * x * x * x * x) : 0.0f;
  float scale = env * mk * 1.4142135623730951f * inv;  // sqrt(2)/safe * env * mask
#pragma unroll
  for (int k = 0; k < NB_B; ++k) {
    float b = scale * __sinf(3.14159265358979f * (float)(k + 1) * safe);
    ef16[(size_t)e * 8 + k] = (_Float16)b;
  }
  maskA[e] = mk;
}

// ---------------- node "up" GEMMs: s_up = s@Wus, v_up[:,:,m] = v[:,:,m]@Wuv ----------------
#define UW 4
__global__ void __launch_bounds__(128) k_node_up(const float* __restrict__ s,
                                                 const float* __restrict__ v,
                                                 const float* __restrict__ Wus,
                                                 const float* __restrict__ Wuv,
                                                 float* __restrict__ s_up,
                                                 float* __restrict__ v_up) {
  __shared__ __align__(16) _Float16 arow[UW][16 * 64];
  const int tid = threadIdx.x, lane = tid & 31, wave = tid >> 5;
  const int half = lane >> 4, nl = lane & 15;
  // Persistent B fragments (f16 converted from f32 global weights; once per kernel)
  v16h Bs[2][4], Bv[2][4];
#pragma unroll
  for (int ks = 0; ks < 2; ++ks)
#pragma unroll
    for (int t = 0; t < 4; ++t) {
      v16h f, g;
#pragma unroll
      for (int e = 0; e < 16; ++e) {
        int k = ks * 32 + 16 * half + e;
        f[e] = (_Float16)Wus[k * 64 + 16 * t + nl];
        g[e] = (_Float16)Wuv[k * 64 + 16 * t + nl];
      }
      Bs[ks][t] = f;
      Bv[ks][t] = g;
    }
  const int ntiles = N_NODES / 16;
  for (int tile = blockIdx.x * UW + wave; tile < ntiles; tile += gridDim.x * UW) {
    const int n0 = tile * 16;
    for (int i = lane; i < 16 * 64; i += 32) arow[wave][i] = (_Float16)s[(size_t)n0 * 64 + i];
    asm volatile("" ::: "memory");
    v16h a0 = load_afrag(arow[wave], 64, 0, lane);
    v16h a1 = load_afrag(arow[wave], 64, 32, lane);
#pragma unroll
    for (int t = 0; t < 4; ++t) {
      v8f c = {};
      c = wmma16(a0, Bs[0][t], c);
      c = wmma16(a1, Bs[1][t], c);
#pragma unroll
      for (int r = 0; r < 8; ++r)
        s_up[((size_t)(n0 + r + 8 * half)) * 64 + 16 * t + nl] = c[r];
    }
    for (int m = 0; m < 3; ++m) {
      asm volatile("" ::: "memory");
      for (int i = lane; i < 16 * 64; i += 32)
        arow[wave][i] = (_Float16)v[((size_t)n0 * 64 + i) * 3 + m];
      asm volatile("" ::: "memory");
      v16h c0 = load_afrag(arow[wave], 64, 0, lane);
      v16h c1 = load_afrag(arow[wave], 64, 32, lane);
#pragma unroll
      for (int t = 0; t < 4; ++t) {
        v8f c = {};
        c = wmma16(c0, Bv[0][t], c);
        c = wmma16(c1, Bv[1][t], c);
#pragma unroll
        for (int r = 0; r < 8; ++r)
          v_up[(((size_t)(n0 + r + 8 * half)) * 64 + 16 * t + nl) * 3 + m] = c[r];
      }
    }
  }
}

// ---------------- fused edge kernel: radial MLP (WMMA) + messages + scatter ----------------
// Weights stored TRANSPOSED in LDS ([n][k]) so every B-fragment load is two ds_load_b128.
#define EW 4
__global__ void __launch_bounds__(128) k_edge(
    const _Float16* __restrict__ ef16, const float* __restrict__ y1g,
    const float* __restrict__ maskA, const int* __restrict__ snd, const int* __restrict__ rcv,
    const float* __restrict__ s_up, const float* __restrict__ v_up,
    const float* __restrict__ Wr1, const float* __restrict__ br1,
    const float* __restrict__ Wr2, const float* __restrict__ br2,
    const float* __restrict__ Wr3, float* __restrict__ agg_s, float* __restrict__ agg_v) {
  __shared__ __align__(16) _Float16 sW1t[64 * 32];    // [n][k], K padded 8 -> 32
  __shared__ __align__(16) _Float16 sW2t[64 * 64];    // [n][k]
  __shared__ __align__(16) _Float16 sW3t[256 * 64];   // [n][k]
  __shared__ float sb1[64], sb2[64];
  __shared__ __align__(16) _Float16 hbuf[EW][16 * 64];  // per-wave h1/h2 staging
  __shared__ float sY[EW][48];
  __shared__ float sM[EW][16];
  __shared__ int sS[EW][16];
  __shared__ int sR[EW][16];

  const int tid = threadIdx.x, lane = tid & 31, wave = tid >> 5;
  const int half = lane >> 4, nl = lane & 15;

  // cooperative f32->f16 transpose of weights into LDS (global reads coalesced)
  for (int i = tid; i < 32 * 64; i += 128) {  // i = k*64 + n, k in [0,32)
    int k = i >> 6, n = i & 63;
    sW1t[n * 32 + k] = (k < NB_B) ? (_Float16)Wr1[i] : (_Float16)0.0f;
  }
  for (int i = tid; i < 64 * 64; i += 128) {  // i = k*64 + n
    int k = i >> 6, n = i & 63;
    sW2t[n * 64 + k] = (_Float16)Wr2[i];
  }
  for (int i = tid; i < 64 * 256; i += 128) {  // i = k*256 + n
    int k = i >> 8, n = i & 255;
    sW3t[n * 64 + k] = (_Float16)Wr3[i];
  }
  if (tid < 64) { sb1[tid] = br1[tid]; sb2[tid] = br2[tid]; }
  __syncthreads();

  const int ntiles = E_EDGES / 16;
  const int tstride = gridDim.x * EW;
  for (int tile = blockIdx.x * EW + wave; tile < ntiles; tile += tstride) {
    const int e0 = tile * 16;
    // prefetch next tile's edge data into cache (global_prefetch_b8)
    const int tnext = tile + tstride;
    if (tnext < ntiles) {
      __builtin_prefetch(ef16 + (size_t)tnext * 16 * 8, 0, 1);
      __builtin_prefetch(y1g + (size_t)tnext * 48, 0, 1);
    }
    if (lane < 16) {
      int e = e0 + lane;
      sS[wave][lane] = snd[e];
      sR[wave][lane] = rcv[e];
      sM[wave][lane] = maskA[e];
      sY[wave][lane * 3 + 0] = y1g[(size_t)e * 3 + 0];
      sY[wave][lane * 3 + 1] = y1g[(size_t)e * 3 + 1];
      sY[wave][lane * 3 + 2] = y1g[(size_t)e * 3 + 2];
    }
    // A fragment of edge_feats: rows = 16 edges, K = 8 (rest zero)
    v16h a1 = {};
    if (lane < 16) {
      const v8h row = *reinterpret_cast<const v8h*>(ef16 + (size_t)(e0 + lane) * 8);
#pragma unroll
      for (int j = 0; j < 8; ++j) a1[j] = row[j];
    }
    // ---- h1 = silu(ef @ Wr1 + br1) ----
#pragma unroll
    for (int t = 0; t < 4; ++t) {
      v16h b = load_bfrag_t(sW1t, 32, 16 * t, 0, lane);
      v8f c = {};
      c = wmma16(a1, b, c);
#pragma unroll
      for (int r = 0; r < 8; ++r) {
        float x = c[r] + sb1[16 * t + nl];
        hbuf[wave][(r + 8 * half) * 64 + 16 * t + nl] = (_Float16)silu_f(x);
      }
    }
    asm volatile("" ::: "memory");
    v16h ah0 = load_afrag(hbuf[wave], 64, 0, lane);
    v16h ah1 = load_afrag(hbuf[wave], 64, 32, lane);
    asm volatile("" ::: "memory");
    // ---- h2 = silu(h1 @ Wr2 + br2) ----
#pragma unroll
    for (int t = 0; t < 4; ++t) {
      v16h b0 = load_bfrag_t(sW2t, 64, 16 * t, 0, lane);
      v16h b1 = load_bfrag_t(sW2t, 64, 16 * t, 32, lane);
      v8f c = {};
      c = wmma16(ah0, b0, c);
      c = wmma16(ah1, b1, c);
#pragma unroll
      for (int r = 0; r < 8; ++r) {
        float x = c[r] + sb2[16 * t + nl];
        hbuf[wave][(r + 8 * half) * 64 + 16 * t + nl] = (_Float16)silu_f(x);
      }
    }
    asm volatile("" ::: "memory");
    v16h a20 = load_afrag(hbuf[wave], 64, 0, lane);
    v16h a21 = load_afrag(hbuf[wave], 64, 32, lane);
    // ---- wts = h2 @ Wr3, then message + scatter straight from accumulators ----
    // column j of Wr3 output maps to (path p = j/64, channel c = j%64)
    for (int tc = 0; tc < 4; ++tc) {
      v8f W[4];
#pragma unroll
      for (int p = 0; p < 4; ++p) {
        int n0 = p * 64 + tc * 16;
        v16h b0 = load_bfrag_t(sW3t, 64, n0, 0, lane);
        v16h b1 = load_bfrag_t(sW3t, 64, n0, 32, lane);
        v8f c = {};
        c = wmma16(a20, b0, c);
        c = wmma16(a21, b1, c);
        W[p] = c;
      }
      const int cch = tc * 16 + nl;  // channel handled by this lane
#pragma unroll
      for (int r = 0; r < 8; ++r) {
        const int m = r + 8 * half;  // edge within tile held at (lane, reg r)
        const float mk = sM[wave][m];
        const int sn = sS[wave][m];
        const int rc = sR[wave][m];
        const float ss = s_up[(size_t)sn * 64 + cch];
        const float* vp = v_up + ((size_t)sn * 64 + cch) * 3;
        const float vv0 = vp[0], vv1 = vp[1], vv2 = vp[2];
        const float yy0 = sY[wave][m * 3 + 0];
        const float yy1 = sY[wave][m * 3 + 1];
        const float yy2 = sY[wave][m * 3 + 2];
        const float w0 = W[0][r] * mk, w1 = W[1][r] * mk;
        const float w2 = W[2][r] * mk, w3 = W[3][r] * mk;
        const float dt = vv0 * yy0 + vv1 * yy1 + vv2 * yy2;
        atomicAdd(&agg_s[(size_t)rc * 64 + cch], w0 * ss + w1 * dt * 0.5773502691896258f);
        float* ap = agg_v + ((size_t)rc * 64 + cch) * 3;
        atomicAdd(ap + 0, w2 * ss * yy0 + w3 * vv0);
        atomicAdd(ap + 1, w2 * ss * yy1 + w3 * vv1);
        atomicAdd(ap + 2, w2 * ss * yy2 + w3 * vv2);
      }
    }
  }
}

// ---------------- node "post": msg GEMMs, species skip, product, readout ----------------
__global__ void __launch_bounds__(256) k_node_post(
    const float* __restrict__ agg_s, const float* __restrict__ agg_v,
    const float* __restrict__ xf, const int* __restrict__ species,
    const float* __restrict__ Wms, const float* __restrict__ Wmv,
    const float* __restrict__ Wscs, const float* __restrict__ Wscv,
    const float* __restrict__ Wps, const float* __restrict__ Wpv,
    const float* __restrict__ Wread, float* __restrict__ s, float* __restrict__ v,
    float* __restrict__ node_acc) {
  __shared__ float A[4][64], T[4][64], S0[4][64], VN[4][64];
  const int tid = threadIdx.x, nb = tid >> 6, c = tid & 63;
  const int n = blockIdx.x * 4 + nb;
  const int sp = species[n];
  const float* Wsc_s = Wscs + (size_t)sp * 64 * 64;
  const float* Wsc_v = Wscv + (size_t)sp * 64 * 64;
  const float xfc = xf[(size_t)n * 64 + c];

  // scalar path
  S0[nb][c] = s[(size_t)n * 64 + c];
  A[nb][c] = agg_s[(size_t)n * 64 + c] * (1.0f / AVGF);
  __syncthreads();
  float msg = 0.0f;
  for (int k = 0; k < 64; ++k) msg += A[nb][k] * Wms[k * 64 + c];
  T[nb][c] = xfc * msg;
  __syncthreads();
  float sn = 0.0f;
  for (int k = 0; k < 64; ++k) sn += T[nb][k] * Wps[k * 64 + c] + S0[nb][k] * Wsc_s[k * 64 + c];
  s[(size_t)n * 64 + c] = sn;

  // vector path, per component
  for (int m = 0; m < 3; ++m) {
    __syncthreads();
    A[nb][c] = agg_v[((size_t)n * 64 + c) * 3 + m] * (1.0f / AVGF);
    S0[nb][c] = v[((size_t)n * 64 + c) * 3 + m];
    __syncthreads();
    float mv = 0.0f;
    for (int k = 0; k < 64; ++k) mv += A[nb][k] * Wmv[k * 64 + c];
    T[nb][c] = xfc * mv;
    __syncthreads();
    float vn = 0.0f;
    for (int k = 0; k < 64; ++k) vn += T[nb][k] * Wpv[k * 64 + c] + S0[nb][k] * Wsc_v[k * 64 + c];
    v[((size_t)n * 64 + c) * 3 + m] = vn;
    VN[nb][c] = vn;
    __syncthreads();
    if (c < 3) {  // o = c : readout out[n,o,m] = sum_k v_new[k]*Wread[o,k]
      float o = 0.0f;
      for (int k = 0; k < 64; ++k) o += VN[nb][k] * Wread[c * 64 + k];
      node_acc[(size_t)n * 9 + c * 3 + m] += silu_f(o);
    }
  }
}

// ---------------- final per-graph reduction ----------------
__global__ void k_reduce(const float* __restrict__ node_acc, float* __restrict__ out) {
  const int g = blockIdx.x, j = blockIdx.y;
  __shared__ float red[256];
  float acc = 0.0f;
  for (int i = threadIdx.x; i < NP_NODES; i += 256)
    acc += node_acc[((size_t)g * NP_NODES + i) * 9 + j];
  red[threadIdx.x] = acc;
  __syncthreads();
  for (int st = 128; st > 0; st >>= 1) {
    if (threadIdx.x < st) red[threadIdx.x] += red[threadIdx.x + st];
    __syncthreads();
  }
  if (threadIdx.x == 0) out[g * 9 + j] = red[0];
}

// ---------------- launch ----------------
extern "C" void kernel_launch(void* const* d_in, const int* in_sizes, int n_in,
                              void* d_out, int out_size, void* d_ws, size_t ws_size,
                              hipStream_t stream) {
  (void)in_sizes; (void)n_in; (void)out_size; (void)ws_size;
  const float* pos     = (const float*)d_in[0];
  const float* cell    = (const float*)d_in[1];
  const float* Sij     = (const float*)d_in[2];
  const float* W_embed = (const float*)d_in[3];
  const float* W_x     = (const float*)d_in[4];
  const float* W_up_s  = (const float*)d_in[5];
  const float* W_up_v  = (const float*)d_in[6];
  const float* Wr1     = (const float*)d_in[7];
  const float* br1     = (const float*)d_in[8];
  const float* Wr2     = (const float*)d_in[9];
  const float* br2     = (const float*)d_in[10];
  const float* Wr3     = (const float*)d_in[11];
  const float* W_msg_s = (const float*)d_in[12];
  const float* W_msg_v = (const float*)d_in[13];
  const float* W_sc_s  = (const float*)d_in[14];
  const float* W_sc_v  = (const float*)d_in[15];
  const float* W_p_s   = (const float*)d_in[16];
  const float* W_p_v   = (const float*)d_in[17];
  const float* W_read  = (const float*)d_in[18];
  const int* edge_index = (const int*)d_in[19];
  const int* batch      = (const int*)d_in[20];
  const int* species    = (const int*)d_in[21];
  const int* snd = edge_index;
  const int* rcv = edge_index + E_EDGES;

  // workspace carve-out (256B aligned slices)
  char* wp = (char*)d_ws;
  auto take = [&wp](size_t bytes) -> char* {
    char* p = wp;
    wp += (bytes + 255) & ~(size_t)255;
    return p;
  };
  _Float16* ef16 = (_Float16*)take((size_t)E_EDGES * 8 * sizeof(_Float16));
  float* y1    = (float*)take((size_t)E_EDGES * 3 * sizeof(float));
  float* maskA = (float*)take((size_t)E_EDGES * sizeof(float));
  float* s     = (float*)take((size_t)N_NODES * 64 * sizeof(float));
  float* xf    = (float*)take((size_t)N_NODES * 64 * sizeof(float));
  float* v     = (float*)take((size_t)N_NODES * 192 * sizeof(float));
  float* s_up  = (float*)take((size_t)N_NODES * 64 * sizeof(float));
  float* v_up  = (float*)take((size_t)N_NODES * 192 * sizeof(float));
  float* agg_s = (float*)take((size_t)N_NODES * 256 * sizeof(float));  // agg_s(64)+agg_v(192)
  float* agg_v = agg_s + (size_t)N_NODES * 64;
  float* node_acc = (float*)take((size_t)N_NODES * 9 * sizeof(float));
  float* M2       = (float*)take((size_t)S_SP * C_CH * sizeof(float));

  k_edge_geom<<<(E_EDGES + 255) / 256, 256, 0, stream>>>(pos, cell, Sij, snd, rcv, batch,
                                                         ef16, y1, maskA);
  k_m2<<<1, 512, 0, stream>>>(W_embed, W_x, M2);
  k_embed<<<(N_NODES * 64 + 255) / 256, 256, 0, stream>>>(species, W_embed, M2, s, xf, v);
  k_zero<<<1024, 256, 0, stream>>>(node_acc, N_NODES * 9);

  for (int l = 0; l < L_LAYERS; ++l) {
    k_node_up<<<313, 128, 0, stream>>>(s, v, W_up_s + (size_t)l * 4096,
                                       W_up_v + (size_t)l * 4096, s_up, v_up);
    k_zero<<<4096, 256, 0, stream>>>(agg_s, N_NODES * 256);
    k_edge<<<1024, 128, 0, stream>>>(ef16, y1, maskA, snd, rcv, s_up, v_up,
                                     Wr1 + (size_t)l * 512, br1 + (size_t)l * 64,
                                     Wr2 + (size_t)l * 4096, br2 + (size_t)l * 64,
                                     Wr3 + (size_t)l * 16384, agg_s, agg_v);
    k_node_post<<<N_NODES / 4, 256, 0, stream>>>(
        agg_s, agg_v, xf, species, W_msg_s + (size_t)l * 4096, W_msg_v + (size_t)l * 4096,
        W_sc_s + (size_t)l * 8 * 4096, W_sc_v + (size_t)l * 8 * 4096,
        W_p_s + (size_t)l * 4096, W_p_v + (size_t)l * 4096, W_read + (size_t)l * 192,
        s, v, node_acc);
  }

  k_reduce<<<dim3(G_GRAPHS, 9), 256, 0, stream>>>(node_acc, (float*)d_out);
}
